// ExpertsChooseContract_25348896981194
// MI455X (gfx1250) — matrix operand order repaired and checked by reading, not compiled, not run on hardware
//
#include <hip/hip_runtime.h>
#include <hip/hip_bf16.h>

// ---- types for WMMA operands (CDNA5 gfx1250, wave32) ----
typedef __attribute__((ext_vector_type(16))) __bf16 v16bf;
typedef __attribute__((ext_vector_type(8)))  __bf16 v8bf;
typedef __attribute__((ext_vector_type(8)))  float  v8f;
typedef __attribute__((ext_vector_type(4)))  float  v4f;

// ---- problem constants (match reference) ----
constexpr int Bv = 4, Tv = 4096, Dv = 1024, Ev = 8, Cv = 1024, Ov = 512;

// ---- tiling ----
constexpr int BM = 128;          // C-rows per workgroup tile
constexpr int BN = 128;          // O-cols per workgroup tile
constexpr int BK = 32;           // K per step (bf16 WMMA K)
constexpr int LDT = BK + 8;      // LDS row pitch in elements (80B -> 20-bank stride, conflict free)
constexpr int KT = Dv / BK;      // 32 K-steps

__global__ __launch_bounds__(256) void moe_wmma_kernel(
    const float* __restrict__ x,      // (B, T, D)
    const int*   __restrict__ eidx,   // (B, E, C)
    const float* __restrict__ w,      // (E, O, D)
    const float* __restrict__ bias,   // (E, O)
    float*       __restrict__ out)    // (B, E, C, O)
{
    __shared__ __bf16 sA[2][BM * LDT];   // gathered tokens tile (M x K), bf16
    __shared__ __bf16 sB[2][BN * LDT];   // weight tile (N x K), bf16
    __shared__ int    sTok[BM];

    const int t  = threadIdx.x;
    const int be = blockIdx.z;           // b*E + e
    const int b  = be / Ev;
    const int e  = be % Ev;
    const int c0 = blockIdx.x * BM;
    const int o0 = blockIdx.y * BN;

    // stage token indices for this tile
    if (t < BM) sTok[t] = eidx[be * Cv + c0 + t];
    __syncthreads();

    // cooperative loader mapping: 256 threads -> 128 rows x 2 sixteen-float segments
    const int r   = t >> 1;
    const int seg = t & 1;
    const float* ap = x + ((size_t)b * Tv + sTok[r]) * Dv + seg * 16;
    const float* bp = w + ((size_t)e * Ov + o0 + r) * Dv + seg * 16;

    // wave / lane mapping for WMMA fragments
    const int lane = t & 31;
    const int wave = t >> 5;
    const int wm   = wave & 3;     // M offset = wm*32
    const int wn   = wave >> 2;    // N offset = wn*64
    const int lr   = lane & 15;
    const int lh   = lane >> 4;    // half-wave selector

    v8f acc[2][4];
#pragma unroll
    for (int m = 0; m < 2; ++m)
#pragma unroll
        for (int n = 0; n < 4; ++n)
#pragma unroll
            for (int i = 0; i < 8; ++i) acc[m][n][i] = 0.0f;

    // convert 16 f32 -> 16 bf16 and store 32B to LDS (two 16B stores)
    auto cvst = [&](__bf16* dst, v4f q0, v4f q1, v4f q2, v4f q3) {
        v8bf p0, p1;
#pragma unroll
        for (int i = 0; i < 4; ++i) {
            p0[i]     = (__bf16)q0[i];
            p0[4 + i] = (__bf16)q1[i];
            p1[i]     = (__bf16)q2[i];
            p1[4 + i] = (__bf16)q3[i];
        }
        *(v8bf*)dst       = p0;
        *(v8bf*)(dst + 8) = p1;
    };

    // ---- preload K-tile 0 into buffer 0 ----
    {
        const v4f* a4 = (const v4f*)ap;
        const v4f* b4 = (const v4f*)bp;
        v4f a0 = a4[0], a1 = a4[1], a2 = a4[2], a3 = a4[3];
        v4f b0 = b4[0], b1 = b4[1], b2 = b4[2], b3 = b4[3];
        cvst(&sA[0][r * LDT + seg * 16], a0, a1, a2, a3);
        cvst(&sB[0][r * LDT + seg * 16], b0, b1, b2, b3);
    }
    __syncthreads();

    // ---- main K loop: prefetch (global->regs), compute (LDS->WMMA), commit (regs->LDS) ----
#pragma unroll 1
    for (int kt = 0; kt < KT; ++kt) {
        const int  cur = kt & 1;
        const bool pre = (kt + 1) < KT;

        v4f pa0, pa1, pa2, pa3, pb0, pb1, pb2, pb3;
        if (pre) {
            const v4f* a4 = (const v4f*)(ap + (size_t)(kt + 1) * BK);
            const v4f* b4 = (const v4f*)(bp + (size_t)(kt + 1) * BK);
            pa0 = a4[0]; pa1 = a4[1]; pa2 = a4[2]; pa3 = a4[3];
            pb0 = b4[0]; pb1 = b4[1]; pb2 = b4[2]; pb3 = b4[3];
        }
        if (kt + 2 < KT) {
            __builtin_prefetch(ap + (size_t)(kt + 2) * BK, 0, 3);  // global_prefetch_b8
            __builtin_prefetch(bp + (size_t)(kt + 2) * BK, 0, 3);
        }

        // A fragments: 16-bit A 16x32 layout — lane holds row (lr), two 8-elem K chunks per half
        v16bf afr[2];
#pragma unroll
        for (int m = 0; m < 2; ++m) {
            const __bf16* s = &sA[cur][(wm * 32 + m * 16 + lr) * LDT + lh * 8];
            v8bf lo = *(const v8bf*)s;
            v8bf hi = *(const v8bf*)(s + 16);
#pragma unroll
            for (int i = 0; i < 8; ++i) { afr[m][i] = lo[i]; afr[m][8 + i] = hi[i]; }
        }

        // B fragments: 16-bit B 32x16 layout — lane holds column (lr), two 8-elem K chunks per half
        v16bf bfr[4];
#pragma unroll
        for (int n = 0; n < 4; ++n) {
            const __bf16* s = &sB[cur][(wn * 64 + n * 16 + lr) * LDT + lh * 8];
            v8bf lo = *(const v8bf*)s;
            v8bf hi = *(const v8bf*)(s + 16);
#pragma unroll
            for (int i = 0; i < 8; ++i) { bfr[n][i] = lo[i]; bfr[n][8 + i] = hi[i]; }
        }

#pragma unroll
        for (int m = 0; m < 2; ++m)
#pragma unroll
            for (int n = 0; n < 4; ++n)
                acc[m][n] = __builtin_amdgcn_wmma_f32_16x16x32_bf16(
                    false, afr[m], false, bfr[n], (short)0, acc[m][n], false, false);

        if (pre) {
            cvst(&sA[cur ^ 1][r * LDT + seg * 16], pa0, pa1, pa2, pa3);
            cvst(&sB[cur ^ 1][r * LDT + seg * 16], pb0, pb1, pb2, pb3);
        }
        __syncthreads();
    }

    // ---- epilogue: bias add + store (C/D layout: VGPR v -> M = base + lh*8 + v, N = lr) ----
    const size_t outBase = (size_t)be * Cv * Ov;
#pragma unroll
    for (int n = 0; n < 4; ++n) {
        const int   N  = o0 + wn * 64 + n * 16 + lr;
        const float bv = bias[e * Ov + N];
#pragma unroll
        for (int m = 0; m < 2; ++m) {
            const int M0 = c0 + wm * 32 + m * 16 + lh * 8;
#pragma unroll
            for (int v = 0; v < 8; ++v) {
                out[outBase + (size_t)(M0 + v) * Ov + N] = acc[m][n][v] + bv;
            }
        }
    }
}

extern "C" void kernel_launch(void* const* d_in, const int* in_sizes, int n_in,
                              void* d_out, int out_size, void* d_ws, size_t ws_size,
                              hipStream_t stream) {
    const float* x    = (const float*)d_in[0];
    const int*   idx  = (const int*)d_in[1];
    const float* w    = (const float*)d_in[2];
    const float* bias = (const float*)d_in[3];
    float*       out  = (float*)d_out;

    dim3 grid(Cv / BM, Ov / BN, Bv * Ev);   // (8, 4, 32)
    dim3 block(256);                        // 8 waves (wave32)
    moe_wmma_kernel<<<grid, block, 0, stream>>>(x, idx, w, bias, out);
}
